// torchKMeans_51951924412425
// MI455X (gfx1250) — matrix operand order repaired and checked by reading, not compiled, not run on hardware
//
#include <hip/hip_runtime.h>
#include <stdint.h>

typedef _Float16 v16h __attribute__((ext_vector_type(16)));
typedef _Float16 v8h  __attribute__((ext_vector_type(8)));
typedef float    v8f  __attribute__((ext_vector_type(8)));

#define N_PTS 262144
#define DIM   128
#define K_CL  1024
#define KC    128                 // clusters per LDS buffer
#define NCHUNK (K_CL / KC)        // 8 chunks
#define ROWB  272                 // padded LDS bytes per cluster row (256B data + 16B pad -> conflict-free)
#define CSQ_OFF (KC * ROWB)       // 34816 (within one buffer)
#define BUFB  (CSQ_OFF + KC * 4)  // 35328 bytes per buffer (16B aligned, 64-DWORD-multiple)
#define SMEM_BYTES (2 * BUFB)     // double buffered: ~70.7 KB
#define MT    2                   // M-tiles (of 16 rows) per wave
#define WG_ROWS (8 * MT * 16)     // 256 rows per 8-wave workgroup

// Generic (flat) pointers into LDS carry the LDS byte offset in their low 32 bits
__device__ __forceinline__ uint32_t lds_addr32(const void* p) {
  return (uint32_t)(uintptr_t)p;
}

__device__ __forceinline__ void async_copy16(uint32_t lds_byte_addr, uint64_t gaddr) {
  // GLOBAL_LOAD_ASYNC_TO_LDS_B128: per-lane 16B global -> LDS, tracked by ASYNCcnt.
  asm volatile("global_load_async_to_lds_b128 %0, %1, off"
               :: "v"(lds_byte_addr), "v"(gaddr)
               : "memory");
}

// ---------------- prep: C (f32) -> C (f16) and c_sq (f32) ----------------
__global__ __launch_bounds__(128) void kmeans_prep(const float* __restrict__ C,
                                                   _Float16* __restrict__ Ch,
                                                   float* __restrict__ cSq) {
  const int k = blockIdx.x;       // one cluster per block
  const int d = threadIdx.x;      // one dim per thread
  const float v = C[k * DIM + d];
  Ch[k * DIM + d] = (_Float16)v;
  float p = v * v;
#pragma unroll
  for (int m = 1; m < 32; m <<= 1) p += __shfl_xor(p, m, 32);
  __shared__ float partial[4];
  if ((threadIdx.x & 31) == 0) partial[threadIdx.x >> 5] = p;
  __syncthreads();
  if (threadIdx.x == 0) cSq[k] = partial[0] + partial[1] + partial[2] + partial[3];
}

// ---------------- main: cross-term WMMA + running argmin ----------------
__global__ __launch_bounds__(256) void kmeans_assign(const float* __restrict__ X,
                                                     const _Float16* __restrict__ Ch,
                                                     const float* __restrict__ cSq,
                                                     int* __restrict__ out) {
  __shared__ __align__(16) unsigned char smem[SMEM_BYTES];

  const int tid   = threadIdx.x;
  const int lane  = tid & 31;
  const int wave  = tid >> 5;
  const int l16   = lane & 15;
  const bool hiL  = (lane >= 16);
  const int rbase = blockIdx.x * WG_ROWS + wave * (MT * 16);  // 32 rows per wave

  const uint32_t lds_base = lds_addr32(smem);

  // ---- stage one 128-cluster chunk (f16 rows + c_sq) into LDS buffer `buf` ----
  auto stage = [&](int cc, int buf) {
    const uint32_t bbase = lds_base + (uint32_t)(buf * BUFB);
#pragma unroll
    for (int i = 0; i < 8; ++i) {
      const int s  = tid + i * 256;        // 0..2047 16B-slots
      const int cl = s >> 4;               // local cluster row
      const int q  = s & 15;               // 16B chunk within row
      const uint32_t laddr = bbase + (uint32_t)(cl * ROWB + q * 16);
      const uint64_t gaddr = (uint64_t)(uintptr_t)(Ch + (size_t)(cc + cl) * DIM) + (uint64_t)(q * 16);
      async_copy16(laddr, gaddr);
    }
    if (tid < KC)
      *(float*)(smem + buf * BUFB + CSQ_OFF + tid * 4) = cSq[cc + tid];
  };

  // ---- Load A fragments (MT tiles of 16 rows x 128 dims as 4 chunks of K=32) ----
  // 16-bit A layout: lane<16 holds row M=lane, K = {0..7, 16..23} (+k0);
  //                  lane>=16 holds row M=lane-16, K = {8..15, 24..31} (+k0).
  v16h afrag[MT][4];
#pragma unroll
  for (int t = 0; t < MT; ++t) {
    const float* xr = X + (size_t)(rbase + t * 16 + l16) * DIM;
    const int d0 = hiL ? 8 : 0;
#pragma unroll
    for (int kc = 0; kc < 4; ++kc) {
      const float4 f0 = *(const float4*)(xr + kc * 32 + d0);
      const float4 f1 = *(const float4*)(xr + kc * 32 + d0 + 4);
      const float4 f2 = *(const float4*)(xr + kc * 32 + d0 + 16);
      const float4 f3 = *(const float4*)(xr + kc * 32 + d0 + 20);
      v16h a;
      a[0]  = (_Float16)f0.x; a[1]  = (_Float16)f0.y; a[2]  = (_Float16)f0.z; a[3]  = (_Float16)f0.w;
      a[4]  = (_Float16)f1.x; a[5]  = (_Float16)f1.y; a[6]  = (_Float16)f1.z; a[7]  = (_Float16)f1.w;
      a[8]  = (_Float16)f2.x; a[9]  = (_Float16)f2.y; a[10] = (_Float16)f2.z; a[11] = (_Float16)f2.w;
      a[12] = (_Float16)f3.x; a[13] = (_Float16)f3.y; a[14] = (_Float16)f3.z; a[15] = (_Float16)f3.w;
      afrag[t][kc] = a;
    }
  }

  float minv[MT][8];
  int   mini[MT][8];
#pragma unroll
  for (int t = 0; t < MT; ++t)
#pragma unroll
    for (int r = 0; r < 8; ++r) { minv[t][r] = 3.402823466e+38f; mini[t][r] = 0; }

  // ---- prologue: stage chunk 0, then pipeline: stage(i+1) || compute(i) ----
  stage(0, 0);
  asm volatile("s_wait_asynccnt 0x0" ::: "memory");
  __syncthreads();

  for (int chunk = 0; chunk < NCHUNK; ++chunk) {
    const int buf = chunk & 1;
    const int cc = chunk * KC;

    // kick off async staging of the NEXT chunk into the other buffer;
    // it overlaps with this chunk's 64 WMMAs per wave.
    if (chunk + 1 < NCHUNK) stage(cc + KC, buf ^ 1);

    const unsigned char* bufp = smem + buf * BUFB;

    // ---- sweep the 128 staged clusters in 16-wide tiles ----
#pragma unroll 2
    for (int c0 = 0; c0 < KC; c0 += 16) {
      // B fragment: lane holds column (cluster) c0 + l16, matching A's K layout:
      // lanes<16 take K {0..7, 16..23} of the chunk, lanes>=16 take K {8..15, 24..31}.
      const unsigned char* bp = bufp + (c0 + l16) * ROWB + (hiL ? 16 : 0);

      // Load ALL four K-chunk B fragments into distinct registers first so the
      // ds_load_b128s pipeline, then run the 2*4 WMMAs back-to-back.
      v16h b[4];
#pragma unroll
      for (int kc = 0; kc < 4; ++kc) {
        const v8h blo = *(const v8h*)(bp + kc * 64);
        const v8h bhi = *(const v8h*)(bp + kc * 64 + 32);
        b[kc] = __builtin_shufflevector(blo, bhi,
            0, 1, 2, 3, 4, 5, 6, 7, 8, 9, 10, 11, 12, 13, 14, 15);
      }

      v8f acc[MT];
#pragma unroll
      for (int t = 0; t < MT; ++t) acc[t] = (v8f){};
#pragma unroll
      for (int kc = 0; kc < 4; ++kc) {
#pragma unroll
        for (int t = 0; t < MT; ++t) {
          acc[t] = __builtin_amdgcn_wmma_f32_16x16x32_f16(
              /*neg_a=*/false, afrag[t][kc], /*neg_b=*/false, b[kc],
              /*c_mod=*/(short)0, acc[t], /*reuse_a=*/false, /*reuse_b=*/false);
        }
      }

      // dist = c_sq - 2*cross  (x_sq dropped: constant per row for argmin)
      const float csq_l = *(const float*)(bufp + CSQ_OFF + (c0 + l16) * 4);
      const int cidx = cc + c0 + l16;
#pragma unroll
      for (int t = 0; t < MT; ++t) {
#pragma unroll
        for (int r = 0; r < 8; ++r) {
          const float d = fmaf(-2.0f, acc[t][r], csq_l);
          if (d < minv[t][r]) { minv[t][r] = d; mini[t][r] = cidx; }
        }
      }
    }

    // next buffer must be fully staged and everyone done reading this one
    asm volatile("s_wait_asynccnt 0x0" ::: "memory");
    __syncthreads();
  }

  // ---- reduce across the 16 columns held by each half-wave ----
#pragma unroll
  for (int m = 1; m < 16; m <<= 1) {
#pragma unroll
    for (int t = 0; t < MT; ++t) {
#pragma unroll
      for (int r = 0; r < 8; ++r) {
        const float ov = __shfl_xor(minv[t][r], m, 32);
        const int   oi = __shfl_xor(mini[t][r], m, 32);
        if (ov < minv[t][r] || (ov == minv[t][r] && oi < mini[t][r])) {
          minv[t][r] = ov; mini[t][r] = oi;
        }
      }
    }
  }
  // lanes<16: rows rbase+t*16+0..7 (acc VGPR r -> M=r);
  // lanes>=16: rows rbase+t*16+8..15 (M=r+8)
  if (l16 == 0) {
#pragma unroll
    for (int t = 0; t < MT; ++t) {
      const int rb = rbase + t * 16 + (hiL ? 8 : 0);
#pragma unroll
      for (int r = 0; r < 8; ++r) out[rb + r] = mini[t][r];
    }
  }
}

// ---------------- launch ----------------
extern "C" void kernel_launch(void* const* d_in, const int* in_sizes, int n_in,
                              void* d_out, int out_size, void* d_ws, size_t ws_size,
                              hipStream_t stream) {
  const float* X = (const float*)d_in[0];            // [N, 128] f32
  const float* C = (const float*)d_in[1];            // [1024, 128] f32
  _Float16* Ch  = (_Float16*)d_ws;                   // 256 KB
  float*    cSq = (float*)((char*)d_ws + (size_t)K_CL * DIM * sizeof(_Float16)); // 4 KB
  int* out = (int*)d_out;                            // [N] int32 labels

  kmeans_prep<<<K_CL, 128, 0, stream>>>(C, Ch, cSq);
  kmeans_assign<<<N_PTS / WG_ROWS, 256, 0, stream>>>(X, Ch, cSq, out);
}
